// PFNLayer_exp_4105988735319
// MI455X (gfx1250) — matrix operand order repaired and checked by reading, not compiled, not run on hardware
//
#include <hip/hip_runtime.h>

#define TPB   256
#define CFEAT 64
#define EPSV  1e-3f

// workspace layout in 32-bit words
#define WS_SUM1 0
#define WS_SQ1  64     // must be WS_SUM1+64
#define WS_S1   128
#define WS_B1   192    // must be WS_S1+64
#define WS_SUM2 256
#define WS_SQ2  320
#define WS_S2   384
#define WS_B2   448
#define WS_FIXED 512   // presence bytes start here

typedef __attribute__((ext_vector_type(16))) __bf16       v16bf;
typedef __attribute__((ext_vector_type(8)))  float        v8f;
typedef __attribute__((ext_vector_type(8)))  unsigned int v8u;

__device__ __forceinline__ unsigned short f2bf(float f) {
  unsigned int u = __float_as_uint(f);
  u += 0x7FFFu + ((u >> 16) & 1u);          // round-to-nearest-even
  return (unsigned short)(u >> 16);
}

// ---------------- K0: zero output + workspace ----------------
__global__ __launch_bounds__(TPB) void k_init(float* __restrict__ out, int out_n,
                                              unsigned int* __restrict__ wsu, int ws_words) {
  int stride = gridDim.x * blockDim.x;
  for (int i = blockIdx.x * blockDim.x + threadIdx.x; i < out_n + ws_words; i += stride) {
    if (i < out_n) out[i] = 0.0f;
    else           wsu[i - out_n] = 0u;
  }
}

// ---------------- K1: BN1 batch stats on raw conv output + presence ----------------
__global__ __launch_bounds__(TPB) void k_stats1(const float* __restrict__ X,
                                                const float* __restrict__ W1,
                                                const int* __restrict__ idx,
                                                float* __restrict__ ws,
                                                unsigned char* __restrict__ present,
                                                int N) {
  __shared__ float sW1[CFEAT * 8];
  __shared__ float sSum[CFEAT], sSq[CFEAT];
  int t = threadIdx.x;
  if (t < CFEAT) { sSum[t] = 0.f; sSq[t] = 0.f; }
  for (int i = t; i < CFEAT * 8; i += TPB) sW1[i] = W1[i];
  __syncthreads();

  int c = t & 63, g = t >> 6;                 // 4 points in flight per iteration
  const int PPB = 2048;
  int base = blockIdx.x * PPB;
  float s = 0.f, q = 0.f;
  for (int it = 0; it < PPB; it += 4) {
    int p = base + it + g;
    if (p < N) {
      const float* xp = X + (size_t)p * 8;
      float y = 0.f;
#pragma unroll
      for (int j = 0; j < 8; ++j) y += xp[j] * sW1[c * 8 + j];
      s += y; q += y * y;
      if (c == 0) present[idx[p]] = 1;        // benign race, same value
    }
  }
  atomicAdd(&sSum[c], s);
  atomicAdd(&sSq[c], q);
  __syncthreads();
  if (t < CFEAT) {
    atomicAdd(&ws[WS_SUM1 + t], sSum[t]);
    atomicAdd(&ws[WS_SQ1 + t], sSq[t]);
  }
}

// ---------------- K2/K4: fold BN into scale/bias ----------------
__global__ void k_finalize(const float* __restrict__ gamma, const float* __restrict__ beta,
                           float* __restrict__ ws, int statOff, int outOff, float invN) {
  int c = threadIdx.x;                        // 64 threads
  float m  = ws[statOff + c] * invN;
  float v  = ws[statOff + 64 + c] * invN - m * m;
  float sc = gamma[c] * rsqrtf(v + EPSV);
  ws[outOff + c]      = sc;
  ws[outOff + 64 + c] = beta[c] - m * sc;
}

// ---------------- K3 (FINAL=0) / K5 (FINAL=1): fused conv+BN1+ReLU+gather-add
//                  then [128,64]x[64,64] bf16 WMMA GEMM ----------------
template <int FINAL>
__global__ __launch_bounds__(TPB) void k_gemm2(const float* __restrict__ X,
                                               const float* __restrict__ SF,
                                               const float* __restrict__ W1,
                                               const float* __restrict__ W2,
                                               const int* __restrict__ idx,
                                               float* __restrict__ ws,
                                               float* __restrict__ out,
                                               int N) {
  __shared__ float          sW1[CFEAT * 8];            // 2 KB
  __shared__ unsigned short sW2[CFEAT * CFEAT];        // 8 KB, bf16, [n][k]
  __shared__ unsigned short sInp[128 * CFEAT];         // 16 KB, bf16, [pt][k]
  __shared__ int            sIdx[128];
  __shared__ float          sBN1[128], sBN2[128];
  __shared__ float          sSum[CFEAT], sSq[CFEAT];

  int t = threadIdx.x;
  int base = blockIdx.x * 128;

  for (int i = t; i < CFEAT * 8; i += TPB) sW1[i] = W1[i];
  for (int i = t; i < CFEAT * CFEAT; i += TPB) sW2[i] = f2bf(W2[i]);
  if (t < 128) {
    sBN1[t] = ws[WS_S1 + t];                           // scale[64] ++ bias[64]
    sBN2[t] = ws[WS_S2 + t];
    int p = base + t;
    sIdx[t] = (p < N) ? idx[p] : 0;
  }
  if (!FINAL && t < CFEAT) { sSum[t] = 0.f; sSq[t] = 0.f; }
  __syncthreads();

  // ---- phase 1: build bf16 input tile: relu(bn1(x@W1^T)) + sparse[idx] ----
  {
    int pt = t >> 1, half = t & 1, c0 = half * 32;
    int p = base + pt;
    unsigned int* inp32 = (unsigned int*)sInp;
    int row32 = (pt * CFEAT + c0) >> 1;
    if (p < N) {
      float xr[8];
#pragma unroll
      for (int j = 0; j < 8; ++j) xr[j] = X[(size_t)p * 8 + j];
      const float* gk = SF + (size_t)sIdx[pt] * CFEAT + c0;
#pragma unroll
      for (int cc = 0; cc < 32; cc += 2) {
        const float* w0 = &sW1[(c0 + cc) * 8];
        float ya = 0.f, yb = 0.f;
#pragma unroll
        for (int j = 0; j < 8; ++j) { ya += xr[j] * w0[j]; yb += xr[j] * w0[8 + j]; }
        int ca = c0 + cc, cb = ca + 1;
        ya = fmaxf(ya * sBN1[ca] + sBN1[64 + ca], 0.f) + gk[cc];
        yb = fmaxf(yb * sBN1[cb] + sBN1[64 + cb], 0.f) + gk[cc + 1];
        inp32[row32 + (cc >> 1)] = (unsigned int)f2bf(ya) | ((unsigned int)f2bf(yb) << 16);
      }
    } else {
#pragma unroll
      for (int cc = 0; cc < 32; cc += 2) inp32[row32 + (cc >> 1)] = 0u;  // zero rows => zero stats
    }
  }
  __syncthreads();

  // ---- phase 2: wave w computes Y[16 pts x 64 ch] with v_wmma_f32_16x16x32_bf16 ----
  int lane = t & 31, w = t >> 5;
  int mrow = lane & 15, hi = lane >> 4;

  // A operands (16x32 bf16 layout per ISA table): lane(m,hi), VGPR v -> k
  const unsigned short* arow = &sInp[(w * 16 + mrow) * CFEAT];
  v8u a0u, a1u;
#pragma unroll
  for (int v = 0; v < 8; ++v) {
    int k = ((v >= 4) ? 16 : 0) + hi * 8 + (v & 3) * 2;
    a0u[v] = *(const unsigned int*)(arow + k);
    a1u[v] = *(const unsigned int*)(arow + k + 32);
  }
  v16bf A0 = __builtin_bit_cast(v16bf, a0u);
  v16bf A1 = __builtin_bit_cast(v16bf, a1u);

#pragma unroll
  for (int tile = 0; tile < 4; ++tile) {
    int n = tile * 16 + (lane & 15);
    // B 32x16 bf16: same k order as A (V0..3: K=8*hi+0..7, V4..7: K=16+8*hi+0..7)
    const unsigned short* brow = &sW2[n * CFEAT];      // B[k][n] = W2[n][k]
    v8u b0u, b1u;
#pragma unroll
    for (int v = 0; v < 8; ++v) {
      int k = ((v >= 4) ? 16 : 0) + hi * 8 + (v & 3) * 2;
      b0u[v] = *(const unsigned int*)(brow + k);
      b1u[v] = *(const unsigned int*)(brow + k + 32);
    }
    v16bf B0 = __builtin_bit_cast(v16bf, b0u);
    v16bf B1 = __builtin_bit_cast(v16bf, b1u);

    v8f acc = {};
    acc = __builtin_amdgcn_wmma_f32_16x16x32_bf16(false, A0, false, B0, (short)0, acc, false, false);
    acc = __builtin_amdgcn_wmma_f32_16x16x32_bf16(false, A1, false, B1, (short)0, acc, false, false);

    if (FINAL) {
      int ch = tile * 16 + (lane & 15);
      float sc = sBN2[ch], bi = sBN2[64 + ch];
#pragma unroll
      for (int r = 0; r < 8; ++r) {
        int m = r + hi * 8;                  // D layout: VGPR r, lane-half hi
        int p = base + w * 16 + m;
        float xv = fmaxf(acc[r] * sc + bi, 0.f);
        if (p < N) {
          int seg = sIdx[w * 16 + m];
          // relu => xv >= 0, so u32 compare == float compare; out pre-zeroed
          atomicMax((unsigned int*)&out[(size_t)seg * CFEAT + ch], __float_as_uint(xv));
        }
      }
    } else {
      float s = 0.f, q = 0.f;
#pragma unroll
      for (int r = 0; r < 8; ++r) { float v_ = acc[r]; s += v_; q += v_ * v_; }
      int ch = tile * 16 + (lane & 15);
      atomicAdd(&sSum[ch], s);
      atomicAdd(&sSq[ch], q);
    }
  }

  if (!FINAL) {
    __syncthreads();
    if (t < CFEAT) {
      atomicAdd(&ws[WS_SUM2 + t], sSum[t]);
      atomicAdd(&ws[WS_SQ2 + t], sSq[t]);
    }
  }
}

// ---------------- K6: absent pillars keep their old sparse features ----------------
__global__ __launch_bounds__(TPB) void k_fixup(const float* __restrict__ SF,
                                               const unsigned char* __restrict__ present,
                                               float* __restrict__ out, int M) {
  int i = blockIdx.x * blockDim.x + threadIdx.x;
  if (i < M * CFEAT) {
    int m = i >> 6;
    if (!present[m]) out[i] = SF[i];
  }
}

extern "C" void kernel_launch(void* const* d_in, const int* in_sizes, int n_in,
                              void* d_out, int out_size, void* d_ws, size_t ws_size,
                              hipStream_t stream) {
  const float* X  = (const float*)d_in[0];   // [N,8]
  const float* SF = (const float*)d_in[1];   // [M,64]
  const float* W1 = (const float*)d_in[2];   // [64,8]
  const float* G1 = (const float*)d_in[3];
  const float* B1 = (const float*)d_in[4];
  const float* W2 = (const float*)d_in[5];   // [64,64]
  const float* G2 = (const float*)d_in[6];
  const float* B2 = (const float*)d_in[7];
  const int*   ID = (const int*)d_in[8];     // [N]

  int N = in_sizes[0] / 8;
  int M = in_sizes[1] / CFEAT;

  float*        out = (float*)d_out;
  float*        ws  = (float*)d_ws;
  unsigned int* wsu = (unsigned int*)d_ws;
  unsigned char* present = (unsigned char*)(wsu + WS_FIXED);
  int presentWords = (M + 3) / 4;            // M bytes, rounded to words
  int wsWords = WS_FIXED + presentWords;
  float invN = 1.0f / (float)N;

  int initTotal = M * CFEAT + wsWords;
  k_init<<<(initTotal + TPB - 1) / TPB, TPB, 0, stream>>>(out, M * CFEAT, wsu, wsWords);

  k_stats1<<<(N + 2047) / 2048, TPB, 0, stream>>>(X, W1, ID, ws, present, N);
  k_finalize<<<1, 64, 0, stream>>>(G1, B1, ws, WS_SUM1, WS_S1, invN);

  int gBlocks = (N + 127) / 128;
  k_gemm2<0><<<gBlocks, TPB, 0, stream>>>(X, SF, W1, W2, ID, ws, out, N);
  k_finalize<<<1, 64, 0, stream>>>(G2, B2, ws, WS_SUM2, WS_S2, invN);
  k_gemm2<1><<<gBlocks, TPB, 0, stream>>>(X, SF, W1, W2, ID, ws, out, N);

  k_fixup<<<(M * CFEAT + TPB - 1) / TPB, TPB, 0, stream>>>(SF, present, out, M);
}
